// Multi_head_self_attention_with_graph_output_1219770712473
// MI455X (gfx1250) — matrix operand order repaired and checked by reading, not compiled, not run on hardware
//
#include <hip/hip_runtime.h>
#include <hip/hip_bf16.h>

// ---------------------------------------------------------------------------
// MI455X (gfx1250) multi-head self-attention with graph (energy) output.
//
// Shapes: B=8, N=1024, E=768, H=8, D=96.
// Outputs: values [B,N,E] f32 (6,291,456) then energy [B,H,N,N] f32
// (67,108,864), concatenated flat in d_out.
//
// Store-bound on the 268MB energy tensor -> single-pass fused
// logits + relative_pos + softmax, bf16 WMMA (v_wmma_f32_16x16x32_bf16)
// for every matmul, Q/K staged as bf16 in workspace.
// ---------------------------------------------------------------------------

typedef __attribute__((ext_vector_type(16))) __bf16 v16bf;
typedef __attribute__((ext_vector_type(8)))  __bf16 v8bf;
typedef __attribute__((ext_vector_type(4)))  __bf16 v4bf;
typedef __attribute__((ext_vector_type(8)))  float  v8f;

union FragA { v16bf v; v8bf h[2]; };

__device__ __forceinline__ __bf16 f2bf(float f) { return (__bf16)f; }

// ---------------------------------------------------------------------------
// f32 -> bf16 conversion, 4 elements/thread (all sizes divisible by 4)
// ---------------------------------------------------------------------------
__global__ __launch_bounds__(256) void cvt_f32_bf16(const float* __restrict__ src,
                                                    __bf16* __restrict__ dst, int n) {
    int i = 4 * (blockIdx.x * blockDim.x + threadIdx.x);
    if (i + 3 < n) {
        float4 f = *(const float4*)(src + i);
        v4bf o;
        o[0] = f2bf(f.x); o[1] = f2bf(f.y); o[2] = f2bf(f.z); o[3] = f2bf(f.w);
        *(v4bf*)(dst + i) = o;
    }
}

// ---------------------------------------------------------------------------
// WMMA fragment loads (per ISA 7.12.2, wave32):
//  A 16x32 bf16: lanes 0-15 row M=lane hold K 0..7 (v0-3) and 16..23 (v4-7);
//                lanes 16-31 row M=lane-16 hold K 8..15 and 24..31.
//  B 32x16 bf16: lane column N=lane&15; lanes 0-15 hold K 0..7 and 16..23,
//                lanes 16-31 hold K 8..15 and 24..31 (same order as A).
// Row-major [row][k] memory with k contiguous feeds both directly.
// ---------------------------------------------------------------------------
__device__ __forceinline__ v16bf load_fragA(const __bf16* __restrict__ base,
                                            int ld, int row0, int k,
                                            int l15, bool hiHalf) {
    const __bf16* p = base + (size_t)(row0 + l15) * ld + k + (hiHalf ? 8 : 0);
    FragA fa;
    fa.h[0] = *(const v8bf*)(p);
    fa.h[1] = *(const v8bf*)(p + 16);
    return fa.v;
}

__device__ __forceinline__ v16bf load_fragB(const __bf16* __restrict__ base,
                                            int ld, int col0, int k,
                                            int l15, bool hiHalf) {
    const __bf16* p = base + (size_t)(col0 + l15) * ld + k + (hiHalf ? 8 : 0);
    FragA fb;
    fb.h[0] = *(const v8bf*)(p);
    fb.h[1] = *(const v8bf*)(p + 16);
    return fb.v;
}

// ---------------------------------------------------------------------------
// GEMM: C[m,n] = sum_k A[m,k]*W[n,k] + bias[n], M=8192, K=768.
// Block: 256 thr (8 waves). Workgroup tile 128(M) x 64(N); each wave 16x64.
// mode 0: f32 store to outF[m*Nout+n]   (values projection)
// mode 1: scatter epilogue per qkv reshape (H,D,2): n = h*192 + d*2 + s,
//         s==0 -> Qd[((b*8+h)*1024+nrow)*96+d], s==1 -> Kd[...], as bf16.
// ---------------------------------------------------------------------------
__global__ __launch_bounds__(256) void gemm_bf16(const __bf16* __restrict__ A,
                                                 const __bf16* __restrict__ W,
                                                 const float* __restrict__ bias,
                                                 float* __restrict__ outF,
                                                 __bf16* __restrict__ Qd,
                                                 __bf16* __restrict__ Kd,
                                                 int Nout, int mode) {
    const int K = 768;
    const int wid  = threadIdx.x >> 5;
    const int lane = threadIdx.x & 31;
    const int l15  = lane & 15;
    const bool hiH = lane >= 16;
    const int m0 = blockIdx.x * 128 + wid * 16;
    const int n0 = blockIdx.y * 64;

    v8f acc[4] = {};
    for (int k = 0; k < K; k += 32) {
        v16bf fa = load_fragA(A, K, m0, k, l15, hiH);
#pragma unroll
        for (int t = 0; t < 4; ++t) {
            v16bf fb = load_fragB(W, K, n0 + t * 16, k, l15, hiH);
            acc[t] = __builtin_amdgcn_wmma_f32_16x16x32_bf16(
                false, fa, false, fb, (short)0, acc[t], false, false);
        }
    }

    // C/D layout: VGPR r, lanes 0-15 -> (M=r, N=lane); lanes 16-31 -> (M=8+r).
#pragma unroll
    for (int t = 0; t < 4; ++t) {
        const int c  = n0 + t * 16 + l15;
        const float bv = bias[c];
#pragma unroll
        for (int r = 0; r < 8; ++r) {
            const int m = m0 + r + (hiH ? 8 : 0);
            const float val = acc[t][r] + bv;
            if (mode == 0) {
                outF[(size_t)m * Nout + c] = val;
            } else {
                const int h = c / 192, rr = c % 192, d = rr >> 1;
                const int b = m >> 10, nrow = m & 1023;
                const size_t idx = (((size_t)(b * 8 + h) * 1024) + nrow) * 96 + d;
                const __bf16 bf = f2bf(val);
                if (rr & 1) Kd[idx] = bf; else Qd[idx] = bf;
            }
        }
    }
}

// ---------------------------------------------------------------------------
// Fused attention energy: per block (bh, rowblock) compute 16 query rows x
// all 1024 key cols. 8 waves, wave w owns cols [w*128, w*128+128) as eight
// 16x16 WMMA tiles (3 K-steps over D=96). Then logits = qk/sqrt(E)+relpos,
// register-resident softmax (width-16 shuffles + 16x8 LDS partials),
// single coalesced f32 store of the 268MB energy tensor.
// ---------------------------------------------------------------------------
__global__ __launch_bounds__(256) void attn_energy(const __bf16* __restrict__ Q,
                                                   const __bf16* __restrict__ Kt,
                                                   const float* __restrict__ relpos,
                                                   float* __restrict__ out) {
    const int N = 1024, D = 96;
    const int bh = blockIdx.x;          // b*8 + h
    const int h  = bh & 7;
    const int i0 = blockIdx.y * 16;     // query row block
    const int wid  = threadIdx.x >> 5;
    const int lane = threadIdx.x & 31;
    const int l15  = lane & 15;
    const bool hiH = lane >= 16;
    const int c0 = wid * 128;

    const __bf16* Qb = Q  + (size_t)bh * N * D;
    const __bf16* Kb = Kt + (size_t)bh * N * D;

    v8f acc[8] = {};
#pragma unroll
    for (int k = 0; k < D; k += 32) {
        v16bf fa = load_fragA(Qb, D, i0, k, l15, hiH);
#pragma unroll
        for (int t = 0; t < 8; ++t) {
            v16bf fb = load_fragB(Kb, D, c0 + t * 16, k, l15, hiH);
            acc[t] = __builtin_amdgcn_wmma_f32_16x16x32_bf16(
                false, fa, false, fb, (short)0, acc[t], false, false);
        }
    }

    // logits: scale by 1/sqrt(768) then add relative_pos[h,row,col]
    const float scale = 0.03608439182435161f;
    float lg[8][8];
#pragma unroll
    for (int t = 0; t < 8; ++t) {
        const int col = c0 + t * 16 + l15;
#pragma unroll
        for (int r = 0; r < 8; ++r) {
            const int row = i0 + r + (hiH ? 8 : 0);
            lg[t][r] = acc[t][r] * scale + relpos[((size_t)h * N + row) * N + col];
        }
    }

    __shared__ float red[16][8];
    __shared__ float fin[16];

    // ---- row max ----
#pragma unroll
    for (int r = 0; r < 8; ++r) {
        float v = lg[0][r];
#pragma unroll
        for (int t = 1; t < 8; ++t) v = fmaxf(v, lg[t][r]);
#pragma unroll
        for (int s = 8; s >= 1; s >>= 1) v = fmaxf(v, __shfl_xor(v, s, 16));
        if (l15 == 0) red[r + (hiH ? 8 : 0)][wid] = v;
    }
    __syncthreads();
    if (threadIdx.x < 16) {
        float v = red[threadIdx.x][0];
        for (int w = 1; w < 8; ++w) v = fmaxf(v, red[threadIdx.x][w]);
        fin[threadIdx.x] = v;
    }
    __syncthreads();
    float rmax[8];
#pragma unroll
    for (int r = 0; r < 8; ++r) rmax[r] = fin[r + (hiH ? 8 : 0)];
    __syncthreads();

    // ---- exp + row sum ----
#pragma unroll
    for (int r = 0; r < 8; ++r) {
        float s = 0.f;
#pragma unroll
        for (int t = 0; t < 8; ++t) {
            float e = __expf(lg[t][r] - rmax[r]);
            lg[t][r] = e;
            s += e;
        }
#pragma unroll
        for (int sh = 8; sh >= 1; sh >>= 1) s += __shfl_xor(s, sh, 16);
        if (l15 == 0) red[r + (hiH ? 8 : 0)][wid] = s;
    }
    __syncthreads();
    if (threadIdx.x < 16) {
        float v = 0.f;
        for (int w = 0; w < 8; ++w) v += red[threadIdx.x][w];
        fin[threadIdx.x] = v;
    }
    __syncthreads();

    // ---- normalize + store (lanes 0-15 / 16-31 each write 64B segments) ----
#pragma unroll
    for (int r = 0; r < 8; ++r) {
        const float inv = 1.0f / fin[r + (hiH ? 8 : 0)];
        const int row = i0 + r + (hiH ? 8 : 0);
        float* op = out + ((size_t)bh * N + row) * N;
#pragma unroll
        for (int t = 0; t < 8; ++t)
            op[c0 + t * 16 + l15] = lg[t][r] * inv;
    }
}

// ---------------------------------------------------------------------------
// Launch
// ---------------------------------------------------------------------------
extern "C" void kernel_launch(void* const* d_in, const int* in_sizes, int n_in,
                              void* d_out, int out_size, void* d_ws, size_t ws_size,
                              hipStream_t stream) {
    (void)in_sizes; (void)n_in; (void)out_size; (void)ws_size;
    const float* x      = (const float*)d_in[0];   // [8,1024,768]
    const float* relpos = (const float*)d_in[1];   // [8,1024,1024]
    const float* qkv_w  = (const float*)d_in[2];   // [1536,768]
    const float* qkv_b  = (const float*)d_in[3];   // [1536]
    const float* val_w  = (const float*)d_in[4];   // [768,768]
    const float* val_b  = (const float*)d_in[5];   // [768]

    float* out_values = (float*)d_out;                         // 6,291,456
    float* out_energy = out_values + (size_t)8 * 1024 * 768;   // 67,108,864

    // workspace (bf16), all sub-buffers 32B-aligned
    __bf16* x_bf    = (__bf16*)d_ws;            // 6,291,456
    __bf16* wqkv_bf = x_bf    + 6291456;        // 1,179,648
    __bf16* wval_bf = wqkv_bf + 1179648;        //   589,824
    __bf16* Qd      = wval_bf + 589824;         // 6,291,456  [b,h,n,d]
    __bf16* Kd      = Qd      + 6291456;        // 6,291,456  [b,h,n,d]

    cvt_f32_bf16<<<6144, 256, 0, stream>>>(x,     x_bf,    6291456);
    cvt_f32_bf16<<<1152, 256, 0, stream>>>(qkv_w, wqkv_bf, 1179648);
    cvt_f32_bf16<<< 576, 256, 0, stream>>>(val_w, wval_bf, 589824);

    // QKV projection, scatter to Q/K bf16
    gemm_bf16<<<dim3(64, 24), 256, 0, stream>>>(x_bf, wqkv_bf, qkv_b,
                                                nullptr, Qd, Kd, 1536, 1);
    // values projection, f32 out
    gemm_bf16<<<dim3(64, 12), 256, 0, stream>>>(x_bf, wval_bf, val_b,
                                                out_values, nullptr, nullptr, 768, 0);
    // fused energy + softmax
    attn_energy<<<dim3(64, 64), 256, 0, stream>>>(Qd, Kd, relpos, out_energy);
}
